// BiLSTM_CRF_40432822124944
// MI455X (gfx1250) — matrix operand order repaired and checked by reading, not compiled, not run on hardware
//
#include <hip/hip_runtime.h>

#define S 2048
#define E 512
#define HD 512
#define G4 2048          // 4*HD
#define NCOL 4096        // forward + backward gate columns
#define T_TAGS 12
#define START_TAG 10
#define STOP_TAG 11
#define NEG_VAL -10000.0f
#define NWG 8            // workgroups per direction in the recurrent phase
#define WSTRIDE 516      // padded bf16 row stride in LDS (8B aligned, bank-conflict-free b64)

typedef __attribute__((ext_vector_type(16))) __bf16 v16bf;
typedef __attribute__((ext_vector_type(8)))  __bf16 v8bf;
typedef __attribute__((ext_vector_type(8)))  float  v8f;
typedef int v2i_t __attribute__((ext_vector_type(2)));

// ---------- helpers ----------
__device__ __forceinline__ unsigned short f2bf(float f) {
    union { float f; unsigned u; } v; v.f = f;
    unsigned r = v.u + 0x7FFFu + ((v.u >> 16) & 1u);   // RNE
    return (unsigned short)(r >> 16);
}
__device__ __forceinline__ float bf_lo(unsigned w) { return __uint_as_float(w << 16); }
__device__ __forceinline__ float bf_hi(unsigned w) { return __uint_as_float(w & 0xFFFF0000u); }
__device__ __forceinline__ float sigmoidf(float x) { return 1.0f / (1.0f + expf(-x)); }

// ---------- init: zero the inter-workgroup barrier counters ----------
__global__ void k_init(int* __restrict__ bar) {
    if (threadIdx.x < 2) bar[threadIdx.x] = 0;
}

// ---------- prep: weights -> bf16 ----------
__global__ void k_prep_w(const float* __restrict__ wih_f, const float* __restrict__ wih_b,
                         const float* __restrict__ whh_f, const float* __restrict__ whh_b,
                         unsigned short* __restrict__ w_all,   // [4096,512] bf16
                         unsigned short* __restrict__ whhF,    // [2048,512] bf16
                         unsigned short* __restrict__ whhB) {
    int i = blockIdx.x * blockDim.x + threadIdx.x;     // over 2048*512
    if (i >= G4 * E) return;
    w_all[i]          = f2bf(wih_f[i]);
    w_all[G4 * E + i] = f2bf(wih_b[i]);
    whhF[i]           = f2bf(whh_f[i]);
    whhB[i]           = f2bf(whh_b[i]);
}

__global__ void k_prep_bias(const float* __restrict__ bih_f, const float* __restrict__ bhh_f,
                            const float* __restrict__ bih_b, const float* __restrict__ bhh_b,
                            float* __restrict__ bias_all) {
    int j = blockIdx.x * blockDim.x + threadIdx.x;
    if (j >= G4) return;
    bias_all[j]      = bih_f[j] + bhh_f[j];
    bias_all[G4 + j] = bih_b[j] + bhh_b[j];
}

// ---------- prep: embedding gather -> bf16 ----------
__global__ void k_embed(const int* __restrict__ sentence, const float* __restrict__ embed,
                        unsigned short* __restrict__ emb_bf) {
    int i = blockIdx.x * blockDim.x + threadIdx.x;     // over 2048*512
    if (i >= S * E) return;
    int s = i >> 9, e = i & (E - 1);
    emb_bf[i] = f2bf(embed[(long long)sentence[s] * E + e]);
}

// ---------- WMMA GEMM:  X[S,4096] = emb[S,512] @ W_all^T + bias ----------
// Each wave computes a 32x64 macro-tile (2 M-tiles x 4 N-tiles): per K-step
// 12 b128 loads feed 8 v_wmma_f32_16x16x32_bf16 (A reused 4x, B reused 2x).
__global__ __launch_bounds__(256) void k_gemm_x(const unsigned short* __restrict__ A_,
                                                const unsigned short* __restrict__ W_,
                                                const float* __restrict__ bias,
                                                float* __restrict__ X) {
    const int lane = threadIdx.x & 31;
    const int wave = threadIdx.x >> 5;
    const int wid  = blockIdx.x * 8 + wave;            // 4096 macro-tiles
    const int tN   = (wid & 63) << 6;                  // 64-col macro base
    const int tM   = (wid >> 6) << 5;                  // 32-row macro base
    const int half = lane >> 4;                        // K-half select
    const int l16  = lane & 15;

    const __bf16* A = (const __bf16*)A_;
    const __bf16* W = (const __bf16*)W_;
    const __bf16* arow0 = A + (size_t)(tM + l16) * E;
    const __bf16* arow1 = A + (size_t)(tM + 16 + l16) * E;
    const __bf16* brow  = W + (size_t)(tN + l16) * E;

    const int aOff = half * 16;    // lanes>=16 hold K=16..31 (matches B layout)
    const int bOff = half * 16;    // lanes>=16 hold K=16..31 (B layout)

    v8f acc[2][4] = {};
    #pragma unroll 4
    for (int k0 = 0; k0 < E; k0 += 32) {
        union { v16bf v; v8bf h[2]; } ua0, ua1;
        ua0.h[0] = *(const v8bf*)(arow0 + k0 + aOff);
        ua0.h[1] = *(const v8bf*)(arow0 + k0 + 8 + aOff);
        ua1.h[0] = *(const v8bf*)(arow1 + k0 + aOff);
        ua1.h[1] = *(const v8bf*)(arow1 + k0 + 8 + aOff);
        #pragma unroll
        for (int j = 0; j < 4; ++j) {
            v16bf b = *(const v16bf*)(brow + (size_t)j * 16 * E + k0 + bOff);
            acc[0][j] = __builtin_amdgcn_wmma_f32_16x16x32_bf16(
                            false, ua0.v, false, b, (short)0, acc[0][j], false, false);
            acc[1][j] = __builtin_amdgcn_wmma_f32_16x16x32_bf16(
                            false, ua1.v, false, b, (short)0, acc[1][j], false, false);
        }
    }

    #pragma unroll
    for (int j = 0; j < 4; ++j) {
        const int col  = tN + j * 16 + l16;
        const float bv = bias[col];
        #pragma unroll
        for (int i = 0; i < 2; ++i) {
            const int rbase = tM + i * 16 + half * 8;
            #pragma unroll
            for (int r = 0; r < 8; ++r)
                X[(size_t)(rbase + r) * NCOL + col] = acc[i][j][r] + bv;
        }
    }
}

// ---------- recurrent LSTM: 8 WGs per direction, Whh slice LDS-resident ----------
// WG `wg` owns hidden units [wg*64, wg*64+64) and their 4 gate rows (256 rows,
// 256x516 bf16 = 258KB LDS, only possible with CDNA5's 320KB). Steps are
// synchronized across the 8 WGs with an atomic phase barrier; h is exchanged
// through L2 via the H output array itself.
__global__ __launch_bounds__(256) void k_lstm(const float* __restrict__ Xpre,      // [S,4096]
                                              const unsigned short* __restrict__ whhF,
                                              const unsigned short* __restrict__ whhB,
                                              const float* __restrict__ h0,
                                              const float* __restrict__ c0,
                                              float* __restrict__ Hf,
                                              float* __restrict__ Hb,
                                              int* __restrict__ bar) {
    extern __shared__ unsigned short whh_s[];          // [256][WSTRIDE] bf16
    __shared__ float sh_h[HD];
    __shared__ float sh_g[256];

    const int tid = threadIdx.x;
    const int dir = blockIdx.x / NWG;                  // 0 = forward, 1 = backward
    const int wg  = blockIdx.x % NWG;
    const unsigned short* Whh = dir ? whhB : whhF;
    float* H = dir ? Hb : Hf;
    const int colofs = dir * G4;
    const int u0 = wg * (HD / NWG);                    // first hidden unit of this WG
    const int g  = tid >> 6;                           // gate 0..3 (i,f,g,o)
    const int j  = tid & 63;                           // unit within slice
    const int grow = g * HD + u0 + j;                  // row in Whh / gate vector

    // stage this WG's 256 Whh rows into LDS (each thread copies its own row)
    {
        const unsigned short* src = Whh + (size_t)grow * HD;
        unsigned short* dst = whh_s + (size_t)tid * WSTRIDE;
#if __has_builtin(__builtin_amdgcn_global_load_async_to_lds_b64) && __has_builtin(__builtin_amdgcn_s_wait_asynccnt)
        for (int k = 0; k < HD; k += 4)
            __builtin_amdgcn_global_load_async_to_lds_b64(
                (__attribute__((address_space(1))) v2i_t*)(src + k),
                (__attribute__((address_space(3))) v2i_t*)(dst + k),
                0, 0);
        __builtin_amdgcn_s_wait_asynccnt(0);
#else
        for (int k = 0; k < HD; k += 4)
            *(uint2*)(dst + k) = *(const uint2*)(src + k);
#endif
    }
    float c_reg = (tid < HD / NWG) ? c0[dir * HD + u0 + tid] : 0.0f;
    __syncthreads();

    const unsigned short* wrow = whh_s + (size_t)tid * WSTRIDE;
    for (int t = 0; t < S; ++t) {
        const int row  = dir ? (S - 1 - t) : t;
        const int prev = dir ? (row + 1) : (row - 1);

        // gather h_{t-1}: 512 floats, 2 per thread
        sh_h[tid]       = (t == 0) ? h0[dir * HD + tid]
                                   : H[(size_t)prev * HD + tid];
        sh_h[256 + tid] = (t == 0) ? h0[dir * HD + 256 + tid]
                                   : H[(size_t)prev * HD + 256 + tid];
        // speculative prefetch of the next step's precomputed gate value
        {
            const int nrow = dir ? (row > 0 ? row - 1 : 0)
                                 : (row < S - 1 ? row + 1 : row);
            __builtin_prefetch(Xpre + (size_t)nrow * NCOL + colofs + grow, 0, 3);
        }
        __syncthreads();

        float acc = Xpre[(size_t)row * NCOL + colofs + grow];
        #pragma unroll 8
        for (int k = 0; k < HD; k += 4) {              // conflict-free ds_read_b64
            uint2 w2 = *(const uint2*)(wrow + k);
            acc = fmaf(sh_h[k + 0], bf_lo(w2.x), acc);
            acc = fmaf(sh_h[k + 1], bf_hi(w2.x), acc);
            acc = fmaf(sh_h[k + 2], bf_lo(w2.y), acc);
            acc = fmaf(sh_h[k + 3], bf_hi(w2.y), acc);
        }
        sh_g[tid] = acc;
        __syncthreads();

        if (tid < HD / NWG) {
            float gi = sigmoidf(sh_g[tid]);            // gate order (i, f, g, o)
            float gf = sigmoidf(sh_g[64 + tid]);
            float gg = tanhf(sh_g[128 + tid]);
            float go = sigmoidf(sh_g[192 + tid]);
            c_reg = gf * c_reg + gi * gg;
            float h = go * tanhf(c_reg);
            H[(size_t)row * HD + u0 + tid] = h;
            __threadfence();
        }
        __syncthreads();

        // cross-workgroup phase barrier (per direction)
        if (tid == 0) {
            __hip_atomic_fetch_add(&bar[dir], 1, __ATOMIC_RELEASE, __HIP_MEMORY_SCOPE_AGENT);
            const int target = NWG * (t + 1);
            while (__hip_atomic_load(&bar[dir], __ATOMIC_ACQUIRE, __HIP_MEMORY_SCOPE_AGENT) < target)
                __builtin_amdgcn_s_sleep(1);
        }
        __syncthreads();
    }
}

// ---------- feats = [Hf|Hb] @ W_out^T + b_out ----------
__global__ void k_feats(const float* __restrict__ Hf, const float* __restrict__ Hb,
                        const float* __restrict__ W_out, const float* __restrict__ b_out,
                        float* __restrict__ feats) {
    int i = blockIdx.x * blockDim.x + threadIdx.x;     // over 2048*12
    if (i >= S * T_TAGS) return;
    int s = i / T_TAGS, t = i - s * T_TAGS;
    const float* hf = Hf + (long long)s * HD;
    const float* hb = Hb + (long long)s * HD;
    const float* wr = W_out + (long long)t * (2 * HD);
    float acc = b_out[t];
    #pragma unroll 4
    for (int k = 0; k < HD; k += 4) {
        float4 h4 = *(const float4*)(hf + k);
        float4 w4 = *(const float4*)(wr + k);
        acc = fmaf(h4.x, w4.x, acc); acc = fmaf(h4.y, w4.y, acc);
        acc = fmaf(h4.z, w4.z, acc); acc = fmaf(h4.w, w4.w, acc);
    }
    #pragma unroll 4
    for (int k = 0; k < HD; k += 4) {
        float4 h4 = *(const float4*)(hb + k);
        float4 w4 = *(const float4*)(wr + HD + k);
        acc = fmaf(h4.x, w4.x, acc); acc = fmaf(h4.y, w4.y, acc);
        acc = fmaf(h4.z, w4.z, acc); acc = fmaf(h4.w, w4.w, acc);
    }
    feats[i] = acc;
}

// ---------- Viterbi decode (single wave) ----------
__global__ void k_viterbi(const float* __restrict__ feats, const float* __restrict__ trans,
                          int* __restrict__ bptrs, float* __restrict__ out) {
    __shared__ float fv[T_TAGS], nfv[T_TAGS];
    const int t = threadIdx.x;
    if (t < T_TAGS) fv[t] = (t == START_TAG) ? 0.0f : NEG_VAL;
    __syncthreads();
    for (int s = 0; s < S; ++s) {
        if (t < T_TAGS) {
            float best = -3.4e38f; int bi = 0;
            #pragma unroll
            for (int p = 0; p < T_TAGS; ++p) {
                float sc = fv[p] + trans[t * T_TAGS + p];
                if (sc > best) { best = sc; bi = p; }
            }
            nfv[t] = best + feats[s * T_TAGS + t];
            bptrs[s * T_TAGS + t] = bi;
        }
        __syncthreads();
        if (t < T_TAGS) fv[t] = nfv[t];
        __syncthreads();
    }
    if (t == 0) {
        float best = -3.4e38f; int bi = 0;
        for (int p = 0; p < T_TAGS; ++p) {
            float sc = fv[p] + trans[STOP_TAG * T_TAGS + p];
            if (sc > best) { best = sc; bi = p; }
        }
        out[0] = best;
        int tag = bi;
        for (int s = S - 1; s >= 0; --s) {
            out[1 + s] = (float)tag;
            tag = bptrs[s * T_TAGS + tag];
        }
    }
}

extern "C" void kernel_launch(void* const* d_in, const int* in_sizes, int n_in,
                              void* d_out, int out_size, void* d_ws, size_t ws_size,
                              hipStream_t stream) {
    const int*   sentence = (const int*)  d_in[0];
    const float* embed    = (const float*)d_in[1];
    const float* wih_f    = (const float*)d_in[2];
    const float* whh_f    = (const float*)d_in[3];
    const float* bih_f    = (const float*)d_in[4];
    const float* bhh_f    = (const float*)d_in[5];
    const float* wih_b    = (const float*)d_in[6];
    const float* whh_b    = (const float*)d_in[7];
    const float* bih_b    = (const float*)d_in[8];
    const float* bhh_b    = (const float*)d_in[9];
    const float* h0       = (const float*)d_in[10];
    const float* c0       = (const float*)d_in[11];
    const float* w_out    = (const float*)d_in[12];
    const float* b_out    = (const float*)d_in[13];
    const float* trans    = (const float*)d_in[14];
    float* out = (float*)d_out;

    // workspace carve-up (all offsets 256B aligned)
    char* p = (char*)d_ws;
    size_t off = 0;
    auto take = [&](size_t bytes) { char* q = p + off; off = (off + bytes + 255) & ~(size_t)255; return q; };
    unsigned short* emb_bf   = (unsigned short*)take((size_t)S * E * 2);        // 2 MB
    unsigned short* w_all    = (unsigned short*)take((size_t)NCOL * E * 2);     // 4 MB
    unsigned short* whhF     = (unsigned short*)take((size_t)G4 * HD * 2);      // 2 MB
    unsigned short* whhB     = (unsigned short*)take((size_t)G4 * HD * 2);      // 2 MB
    float*          bias_all = (float*)take((size_t)NCOL * 4);
    float*          X_all    = (float*)take((size_t)S * NCOL * 4);              // 32 MB
    float*          Hf       = (float*)take((size_t)S * HD * 4);                // 4 MB
    float*          Hb       = (float*)take((size_t)S * HD * 4);                // 4 MB
    float*          feats    = (float*)take((size_t)S * T_TAGS * 4);
    int*            bptrs    = (int*)  take((size_t)S * T_TAGS * 4);
    int*            bar      = (int*)  take(256);
    (void)ws_size; (void)in_sizes; (void)n_in; (void)out_size;

    k_init     <<<1, 64, 0, stream>>>(bar);
    k_prep_w   <<<(G4 * E + 255) / 256, 256, 0, stream>>>(wih_f, wih_b, whh_f, whh_b,
                                                          w_all, whhF, whhB);
    k_prep_bias<<<(G4 + 255) / 256, 256, 0, stream>>>(bih_f, bhh_f, bih_b, bhh_b, bias_all);
    k_embed    <<<(S * E + 255) / 256, 256, 0, stream>>>(sentence, embed, emb_bf);

    // 64 M-pairs x 64 N-quads = 4096 wave macro-tiles, 8 waves per block
    k_gemm_x   <<<512, 256, 0, stream>>>(emb_bf, w_all, bias_all, X_all);

    // 2 directions x 8 cooperating WGs, 258KB dynamic LDS each
    k_lstm     <<<2 * NWG, 256, (size_t)256 * WSTRIDE * 2, stream>>>(
                   X_all, whhF, whhB, h0, c0, Hf, Hb, bar);

    k_feats    <<<(S * T_TAGS + 255) / 256, 256, 0, stream>>>(Hf, Hb, w_out, b_out, feats);
    k_viterbi  <<<1, 32, 0, stream>>>(feats, trans, bptrs, out);
}
